// Transformer_10660108828894
// MI455X (gfx1250) — matrix operand order repaired and checked by reading, not compiled, not run on hardware
//
#include <hip/hip_runtime.h>

typedef __attribute__((ext_vector_type(16))) _Float16 v16h;
typedef __attribute__((ext_vector_type(4)))  _Float16 v4h;
typedef __attribute__((ext_vector_type(8)))  float    v8f;

union Frag { v16h v; unsigned u[8]; };

// A-matrix fragment (16xK strip, 16-bit elements), per CDNA5 ISA 7.12.2:
// lane = (k-half)*16 + m ; VGPR i holds K = 16*(i>=4) + 8*(lane>>4) + 2*(i&3) (+1)
__device__ __forceinline__ void loadA(Frag& a, const _Float16* base, int pitch,
                                      int m, int kbase, int lane) {
  const int half = ((lane >> 4) << 3); // 0 or 8
#pragma unroll
  for (int i = 0; i < 8; ++i) {
    const int k = kbase + ((i >> 2) << 4) + half + ((i & 3) << 1);
    a.u[i] = *(const unsigned*)(base + m * pitch + k);
  }
}

// B-matrix fragment (Kx16, 16-bit): lane = (k-half)*16 + n ; VGPR j holds K = 16*(j>=4) + 8*(lane>>4) + 2*(j&3) (+1)
// base is stored TRANSPOSED in LDS as [n][k] so the pair is one 32-bit read.
__device__ __forceinline__ void loadB(Frag& b, const _Float16* base, int pitch,
                                      int n, int kbase, int lane) {
  const int kg = (lane >> 4) << 3; // 0 or 8
#pragma unroll
  for (int j = 0; j < 8; ++j)
    b.u[j] = *(const unsigned*)(base + n * pitch + kbase + ((j >> 2) << 4) + kg + ((j & 3) << 1));
}

// ---------- small GEMM (64x64 tile) for per-head QKV projections ----------
// C = A[M,K] * (W + z*wz)[K,N=64] + bias[z*biasz + n]; out col = z*colz + n.
__global__ __launch_bounds__(128) void gemm_kernel(
    const float* __restrict__ A, int lda,
    const float* __restrict__ W, int ldb, long long wz,
    const float* __restrict__ bias, int biasz,
    float* __restrict__ C, int ldc, int colz, int K)
{
  constexpr int PA = 36;
  __shared__ __align__(16) _Float16 As[64 * PA];
  __shared__ __align__(16) _Float16 Bs[64 * PA];
  const int tid = threadIdx.x, lane = tid & 31, wave = tid >> 5;
  const int m0 = blockIdx.y * 64;
  const int z  = blockIdx.z;
  const float* Wz = W + (long long)z * wz;
  const float* bz = bias + (size_t)z * (size_t)biasz;
  const int c0 = z * colz;

  const v8f vz8 = {0.f,0.f,0.f,0.f,0.f,0.f,0.f,0.f};
  v8f acc[4];
#pragma unroll
  for (int tn = 0; tn < 4; ++tn) acc[tn] = vz8;

  const int am = wave * 16 + (lane & 15);

  for (int kk = 0; kk < K; kk += 32) {
    __syncthreads();
    for (int i = tid; i < 64 * 8; i += 128) {          // A tile [m][k], float4 loads
      int m = i >> 3, q = (i & 7) << 2;
      float4 f = *(const float4*)(A + (size_t)(m0 + m) * lda + kk + q);
      v4h h; h[0] = (_Float16)f.x; h[1] = (_Float16)f.y;
             h[2] = (_Float16)f.z; h[3] = (_Float16)f.w;
      *(v4h*)&As[m * PA + q] = h;
    }
    for (int i = tid; i < 32 * 16; i += 128) {         // B tile -> Bt[n][k]
      int k = i >> 4, q = (i & 15) << 2;
      float4 f = *(const float4*)(Wz + (size_t)(kk + k) * ldb + q);
      Bs[(q + 0) * PA + k] = (_Float16)f.x;
      Bs[(q + 1) * PA + k] = (_Float16)f.y;
      Bs[(q + 2) * PA + k] = (_Float16)f.z;
      Bs[(q + 3) * PA + k] = (_Float16)f.w;
    }
    if (kk + 32 < K) {                                 // global_prefetch_b8 next chunk
      __builtin_prefetch(A + (size_t)(m0 + (tid >> 1)) * lda + kk + 32 + (tid & 1) * 16, 0, 1);
      __builtin_prefetch(Wz + (size_t)(kk + 32 + (tid >> 3)) * ldb + (tid & 7) * 8, 0, 1);
    }
    __syncthreads();
    Frag a; loadA(a, As, PA, am, 0, lane);
#pragma unroll
    for (int tn = 0; tn < 4; ++tn) {
      Frag bb; loadB(bb, Bs, PA, tn * 16 + (lane & 15), 0, lane);
      acc[tn] = __builtin_amdgcn_wmma_f32_16x16x32_f16(
          false, a.v, false, bb.v, (short)0, acc[tn], false, false);
    }
  }

  const int rbase = wave * 16 + ((lane >> 4) << 3);
#pragma unroll
  for (int tn = 0; tn < 4; ++tn) {
#pragma unroll
    for (int r = 0; r < 8; ++r) {
      const int row = m0 + rbase + r;
      const int nn  = tn * 16 + (lane & 15);
      C[(size_t)row * ldc + c0 + nn] = acc[tn][r] + bz[nn];
    }
  }
}

// ---------- big GEMM (128x128 tile, 256 thr / 8 waves) ----------
// wave wr=wave&3 -> 32-row strip; wc=wave>>2 -> 64-col half.
__global__ __launch_bounds__(256) void gemm128_kernel(
    const float* __restrict__ A, int lda,
    const float* __restrict__ W, int ldb,
    const float* __restrict__ bias,
    const float* __restrict__ res,
    float* __restrict__ C, int ldc,
    int K, int relu)
{
  constexpr int PA = 36;
  __shared__ __align__(16) _Float16 As[128 * PA];
  __shared__ __align__(16) _Float16 Bs[128 * PA];
  const int tid = threadIdx.x, lane = tid & 31, wave = tid >> 5;
  const int wr = wave & 3, wc = wave >> 2;
  const int m0 = blockIdx.y * 128, n0 = blockIdx.x * 128;
  const int lm16 = lane & 15;

  const v8f vz8 = {0.f,0.f,0.f,0.f,0.f,0.f,0.f,0.f};
  v8f acc[2][4];
#pragma unroll
  for (int t = 0; t < 2; ++t)
#pragma unroll
    for (int tn = 0; tn < 4; ++tn) acc[t][tn] = vz8;

  const int amBase = 32 * wr + lm16;

  for (int kk = 0; kk < K; kk += 32) {
    __syncthreads();
    for (int i = tid; i < 128 * 8; i += 256) {         // A tile [m][k]
      int m = i >> 3, q = (i & 7) << 2;
      float4 f = *(const float4*)(A + (size_t)(m0 + m) * lda + kk + q);
      v4h h; h[0] = (_Float16)f.x; h[1] = (_Float16)f.y;
             h[2] = (_Float16)f.z; h[3] = (_Float16)f.w;
      *(v4h*)&As[m * PA + q] = h;
    }
    for (int i = tid; i < 32 * 32; i += 256) {         // B tile -> Bt[n][k]
      int k = i >> 5, q = (i & 31) << 2;
      float4 f = *(const float4*)(W + (size_t)(kk + k) * ldb + n0 + q);
      Bs[(q + 0) * PA + k] = (_Float16)f.x;
      Bs[(q + 1) * PA + k] = (_Float16)f.y;
      Bs[(q + 2) * PA + k] = (_Float16)f.z;
      Bs[(q + 3) * PA + k] = (_Float16)f.w;
    }
    if (kk + 32 < K) {                                 // prefetch next K-chunk
      __builtin_prefetch(A + (size_t)(m0 + (tid >> 1)) * lda + kk + 32 + (tid & 1) * 16, 0, 1);
      __builtin_prefetch(W + (size_t)(kk + 32 + (tid >> 3)) * ldb + n0 + (tid & 7) * 16, 0, 1);
    }
    __syncthreads();
    Frag a0, a1;
    loadA(a0, As, PA, amBase,      0, lane);
    loadA(a1, As, PA, amBase + 16, 0, lane);
#pragma unroll
    for (int tn = 0; tn < 4; ++tn) {
      Frag bb; loadB(bb, Bs, PA, 64 * wc + tn * 16 + lm16, 0, lane);
      acc[0][tn] = __builtin_amdgcn_wmma_f32_16x16x32_f16(
          false, a0.v, false, bb.v, (short)0, acc[0][tn], false, false);
      acc[1][tn] = __builtin_amdgcn_wmma_f32_16x16x32_f16(
          false, a1.v, false, bb.v, (short)0, acc[1][tn], false, false);
    }
  }

  const int rsub = (lane >> 4) << 3;
#pragma unroll
  for (int t = 0; t < 2; ++t)
#pragma unroll
    for (int tn = 0; tn < 4; ++tn)
#pragma unroll
      for (int r = 0; r < 8; ++r) {
        const int row = m0 + 32 * wr + 16 * t + rsub + r;
        const int col = n0 + 64 * wc + tn * 16 + lm16;
        float v = acc[t][tn][r] + bias[col];
        if (res)  v += res[(size_t)row * ldc + col];
        if (relu) v = fmaxf(v, 0.f);
        C[(size_t)row * ldc + col] = v;
      }
}

// ---------- flash attention per (b,h) ----------
__global__ __launch_bounds__(128) void attn_kernel(
    const float* __restrict__ Q, const float* __restrict__ K,
    const float* __restrict__ V, float* __restrict__ O,
    const int* __restrict__ mask)
{
  constexpr int Dm = 512, SEQ = 512, P = 68;
  __shared__ __align__(16) _Float16 Qs[64 * P];
  __shared__ __align__(16) _Float16 Ks[64 * P];   // [s][e]  == B^T for Q*K^T
  __shared__ __align__(16) _Float16 Vt[64 * P];   // [e][s]  == B^T for P*V
  __shared__ __align__(16) _Float16 Ps[64 * P];

  const int tid = threadIdx.x, lane = tid & 31, wave = tid >> 5;
  const int b = blockIdx.y >> 3, h = blockIdx.y & 7;
  const int t0 = blockIdx.x * 64;
  const int hc = h * 64;

  for (int i = tid; i < 64 * 64; i += 128) {
    int r = i >> 6, c = i & 63;
    Qs[r * P + c] = (_Float16)Q[(size_t)(b * SEQ + t0 + r) * Dm + hc + c];
  }

  float mrow[8], lsum[8];
  const v8f vz8 = {0.f,0.f,0.f,0.f,0.f,0.f,0.f,0.f};
  v8f oacc[4];
#pragma unroll
  for (int r = 0; r < 8; ++r) { mrow[r] = -1e30f; lsum[r] = 0.f; }
#pragma unroll
  for (int tn = 0; tn < 4; ++tn) oacc[tn] = vz8;

  const int am = wave * 16 + (lane & 15);
  const int rbase = wave * 16 + ((lane >> 4) << 3);

  for (int s0 = 0; s0 < SEQ; s0 += 64) {
    __syncthreads();
    for (int i = tid; i < 64 * 64; i += 128) {
      int r = i >> 6, c = i & 63;
      float kv = K[(size_t)(b * SEQ + s0 + r) * Dm + hc + c];
      float vv = V[(size_t)(b * SEQ + s0 + r) * Dm + hc + c];
      Ks[r * P + c] = (_Float16)kv;
      Vt[c * P + r] = (_Float16)vv;
    }
    __syncthreads();

    v8f sc[4];
#pragma unroll
    for (int tn = 0; tn < 4; ++tn) sc[tn] = vz8;
#pragma unroll
    for (int kc = 0; kc < 64; kc += 32) {
      Frag a; loadA(a, Qs, P, am, kc, lane);
#pragma unroll
      for (int tn = 0; tn < 4; ++tn) {
        Frag bb; loadB(bb, Ks, P, tn * 16 + (lane & 15), kc, lane);
        sc[tn] = __builtin_amdgcn_wmma_f32_16x16x32_f16(
            false, a.v, false, bb.v, (short)0, sc[tn], false, false);
      }
    }

    float madd[4];
#pragma unroll
    for (int tn = 0; tn < 4; ++tn) {
      int sg = s0 + tn * 16 + (lane & 15);
      madd[tn] = (mask[b * SEQ + sg] != 0) ? 0.f : -1e9f;
    }

#pragma unroll
    for (int r = 0; r < 8; ++r) {
      float e0 = sc[0][r] * 0.125f + madd[0];
      float e1 = sc[1][r] * 0.125f + madd[1];
      float e2 = sc[2][r] * 0.125f + madd[2];
      float e3 = sc[3][r] * 0.125f + madd[3];
      float lm = fmaxf(fmaxf(e0, e1), fmaxf(e2, e3));
      for (int o = 1; o <= 8; o <<= 1) lm = fmaxf(lm, __shfl_xor(lm, o, 32));
      float nm = fmaxf(mrow[r], lm);
      float fr = __expf(mrow[r] - nm);
      mrow[r] = nm;
      float p0 = __expf(e0 - nm), p1 = __expf(e1 - nm);
      float p2 = __expf(e2 - nm), p3 = __expf(e3 - nm);
      float ps = p0 + p1 + p2 + p3;
      for (int o = 1; o <= 8; o <<= 1) ps += __shfl_xor(ps, o, 32);
      lsum[r] = lsum[r] * fr + ps;
      const int prow = rbase + r;
      Ps[prow * P +  0 + (lane & 15)] = (_Float16)p0;
      Ps[prow * P + 16 + (lane & 15)] = (_Float16)p1;
      Ps[prow * P + 32 + (lane & 15)] = (_Float16)p2;
      Ps[prow * P + 48 + (lane & 15)] = (_Float16)p3;
#pragma unroll
      for (int tn = 0; tn < 4; ++tn) oacc[tn][r] *= fr;
    }
    __syncthreads();

#pragma unroll
    for (int kc = 0; kc < 64; kc += 32) {
      Frag a; loadA(a, Ps, P, am, kc, lane);
#pragma unroll
      for (int tn = 0; tn < 4; ++tn) {
        Frag bb; loadB(bb, Vt, P, tn * 16 + (lane & 15), kc, lane);
        oacc[tn] = __builtin_amdgcn_wmma_f32_16x16x32_f16(
            false, a.v, false, bb.v, (short)0, oacc[tn], false, false);
      }
    }
  }

#pragma unroll
  for (int tn = 0; tn < 4; ++tn)
#pragma unroll
    for (int r = 0; r < 8; ++r) {
      int row = t0 + rbase + r;
      int col = hc + tn * 16 + (lane & 15);
      O[(size_t)(b * SEQ + row) * Dm + col] = oacc[tn][r] / lsum[r];
    }
}

__global__ __launch_bounds__(256) void ln_kernel(
    const float* __restrict__ X, const float* __restrict__ g,
    const float* __restrict__ bta, float* __restrict__ Y)
{
  __shared__ float red[256];
  const int row = blockIdx.x, tid = threadIdx.x;
  const float* xr = X + (size_t)row * 512;
  float x0 = xr[tid], x1 = xr[tid + 256];
  red[tid] = x0 + x1;
  __syncthreads();
  for (int o = 128; o > 0; o >>= 1) { if (tid < o) red[tid] += red[tid + o]; __syncthreads(); }
  float mu = red[0] * (1.f / 512.f);
  __syncthreads();
  float d0 = x0 - mu, d1 = x1 - mu;
  red[tid] = d0 * d0 + d1 * d1;
  __syncthreads();
  for (int o = 128; o > 0; o >>= 1) { if (tid < o) red[tid] += red[tid + o]; __syncthreads(); }
  float rs = rsqrtf(red[0] * (1.f / 512.f) + 1e-5f);
  float* yr = Y + (size_t)row * 512;
  yr[tid]       = d0 * rs * g[tid]       + bta[tid];
  yr[tid + 256] = d1 * rs * g[tid + 256] + bta[tid + 256];
}

extern "C" void kernel_launch(void* const* d_in, const int* in_sizes, int n_in,
                              void* d_out, int out_size, void* d_ws, size_t ws_size,
                              hipStream_t stream) {
  (void)in_sizes; (void)n_in; (void)out_size; (void)ws_size;
  constexpr int Dm = 512, Hh = 8, DHd = 64, Ff = 2048, Ll = 6;
  constexpr int Bb = 16, Ss = 512;
  const size_t ACT = (size_t)Bb * Ss * Dm;

  const float* src  = (const float*)d_in[0];
  const float* tgt  = (const float*)d_in[1];
  const int*   smask = (const int*)d_in[2];
  const int*   tmask = (const int*)d_in[3];
  auto EP = [&](int j) { return (const float*)d_in[4 + j];  };  // 16 encoder params
  auto DP = [&](int j) { return (const float*)d_in[20 + j]; };  // 26 decoder params

  const size_t WqL = (size_t)Hh * Dm * DHd, bqL = (size_t)Hh * DHd;
  const size_t WoL = (size_t)Dm * Dm, vecL = Dm;
  const size_t W1L = (size_t)Dm * Ff, b1L = Ff, W2L = (size_t)Ff * Dm;

  float* X  = (float*)d_ws;      // encoder act / enc_out
  float* Y  = X  + ACT;          // decoder act
  float* Qb = Y  + ACT;
  float* Kb = Qb + ACT;
  float* Vb = Kb + ACT;
  float* Ob = Vb + ACT;
  float* Tb = Ob + ACT;
  float* H1 = Tb + ACT;          // [8192, 2048]

  hipMemcpyAsync(X, src, ACT * sizeof(float), hipMemcpyDeviceToDevice, stream);
  hipMemcpyAsync(Y, tgt, ACT * sizeof(float), hipMemcpyDeviceToDevice, stream);

  const int MB64  = (Bb * Ss) / 64;   // 128
  const int MB128 = (Bb * Ss) / 128;  // 64

  auto qkv = [&](const float* act, const float* Wb, const float* bb, float* out) {
    gemm_kernel<<<dim3(1, MB64, Hh), dim3(128), 0, stream>>>(
        act, Dm, Wb, DHd, (long long)Dm * DHd, bb, DHd, out, Dm, DHd, Dm);
  };
  auto proj = [&](const float* act, const float* Wb, const float* bb,
                  const float* res, float* out) {
    gemm128_kernel<<<dim3(Dm / 128, MB128), dim3(256), 0, stream>>>(
        act, Dm, Wb, Dm, bb, res, out, Dm, Dm, 0);
  };
  auto ffn1 = [&](const float* act, const float* Wb, const float* bb, float* out) {
    gemm128_kernel<<<dim3(Ff / 128, MB128), dim3(256), 0, stream>>>(
        act, Dm, Wb, Ff, bb, nullptr, out, Ff, Dm, 1);
  };
  auto ffn2 = [&](const float* act, const float* Wb, const float* bb,
                  const float* res, float* out) {
    gemm128_kernel<<<dim3(Dm / 128, MB128), dim3(256), 0, stream>>>(
        act, Ff, Wb, Dm, bb, res, out, Dm, Ff, 0);
  };
  auto attn = [&](const int* mask) {
    attn_kernel<<<dim3(Ss / 64, Bb * Hh), dim3(128), 0, stream>>>(Qb, Kb, Vb, Ob, mask);
  };
  auto ln = [&](const float* in, const float* g, const float* b, float* out) {
    ln_kernel<<<dim3(Bb * Ss), dim3(256), 0, stream>>>(in, g, b, out);
  };

  // ---------------- Encoder ----------------
  for (int i = 0; i < Ll; ++i) {
    qkv(X, EP(0) + i * WqL, EP(1) + i * bqL, Qb);
    qkv(X, EP(2) + i * WqL, EP(3) + i * bqL, Kb);
    qkv(X, EP(4) + i * WqL, EP(5) + i * bqL, Vb);
    attn(smask);
    proj(Ob, EP(6) + i * WoL, EP(7) + i * vecL, X, Tb);   // + residual x
    ln(Tb, EP(8) + i * vecL, EP(9) + i * vecL, X);        // ln1
    ffn1(X, EP(10) + i * W1L, EP(11) + i * b1L, H1);
    ffn2(H1, EP(12) + i * W2L, EP(13) + i * vecL, nullptr, Tb); // NO residual
    ln(Tb, EP(14) + i * vecL, EP(15) + i * vecL, X);      // ln2
  }

  // ---------------- Decoder ----------------
  for (int i = 0; i < Ll; ++i) {
    // self-attention
    qkv(Y, DP(0) + i * WqL, DP(1) + i * bqL, Qb);
    qkv(Y, DP(2) + i * WqL, DP(3) + i * bqL, Kb);
    qkv(Y, DP(4) + i * WqL, DP(5) + i * bqL, Vb);
    attn(tmask);
    proj(Ob, DP(6) + i * WoL, DP(7) + i * vecL, Y, Tb);
    ln(Tb, DP(8) + i * vecL, DP(9) + i * vecL, Y);        // ln1
    // cross-attention (K/V from encoder output X)
    qkv(Y, DP(10) + i * WqL, DP(11) + i * bqL, Qb);
    qkv(X, DP(12) + i * WqL, DP(13) + i * bqL, Kb);
    qkv(X, DP(14) + i * WqL, DP(15) + i * bqL, Vb);
    attn(smask);
    proj(Ob, DP(16) + i * WoL, DP(17) + i * vecL, Y, Tb);
    ln(Tb, DP(18) + i * vecL, DP(19) + i * vecL, Y);      // ln2
    // FFN with residual
    ffn1(Y, DP(20) + i * W1L, DP(21) + i * b1L, H1);
    ffn2(H1, DP(22) + i * W2L, DP(23) + i * vecL, Y, Tb);
    ln(Tb, DP(24) + i * vecL, DP(25) + i * vecL, Y);      // ln3
  }

  hipMemcpyAsync(d_out, Y, ACT * sizeof(float), hipMemcpyDeviceToDevice, stream);
}
